// temporal_2_30683246363007
// MI455X (gfx1250) — matrix operand order repaired and checked by reading, not compiled, not run on hardware
//
#include <hip/hip_runtime.h>

// ---------------------------------------------------------------------------
// Problem constants (match reference)
// ---------------------------------------------------------------------------
#define BB   8
#define LL   5
#define SS   512
#define DD   1024
#define HH   16
#define DHH  64
#define LD_O 5120   // L*H*Dh = L*D

typedef __attribute__((ext_vector_type(16))) __bf16 v16bf;
typedef __attribute__((ext_vector_type(8)))  __bf16 v8bf;
typedef __attribute__((ext_vector_type(8)))  float  v8f;

// Scheduling hints: pin [VMEM-read group] -> [WMMA group] per pipeline stage.
#if defined(__has_builtin)
#if __has_builtin(__builtin_amdgcn_sched_group_barrier)
#define SCHED_GROUP(mask, size, id) __builtin_amdgcn_sched_group_barrier(mask, size, id)
#endif
#endif
#ifndef SCHED_GROUP
#define SCHED_GROUP(mask, size, id)
#endif
#define SG_VMEM_READ 0x020
#define SG_MFMA      0x008

// ---------------------------------------------------------------------------
// Fragment helpers (CDNA5 WMMA 16x16x32 bf16 layouts, wave32)
//
// A (16xK=32):  lane l holds row m=l&15; halfword i: k = (i<8 ? i : 8+i) +
//               (l>=16 ? 8 : 0)  -> two contiguous 8-element runs at +0,+16.
// B (K=32x16):  lane l holds col n=l&15; halfword i: same k as A
//               -> two contiguous 8-element runs at +0,+16.
// C/D (16x16):  vgpr j, lane l: m = j + (l>=16?8:0), n = l&15.
// ---------------------------------------------------------------------------
__device__ __forceinline__ v16bf frag_a(const __bf16* p /* row + k0 + koffA */) {
  v8bf lo = *(const v8bf*)(p);
  v8bf hi = *(const v8bf*)(p + 16);
  return __builtin_shufflevector(lo, hi, 0,1,2,3,4,5,6,7,8,9,10,11,12,13,14,15);
}
__device__ __forceinline__ v16bf frag_b(const __bf16* p /* colrow + k0 + koffB */) {
  v8bf lo = *(const v8bf*)(p);
  v8bf hi = *(const v8bf*)(p + 16);
  return __builtin_shufflevector(lo, hi, 0,1,2,3,4,5,6,7,8,9,10,11,12,13,14,15);
}
__device__ __forceinline__ v8f wmma_bf16(v16bf a, v16bf b, v8f c) {
  return __builtin_amdgcn_wmma_f32_16x16x32_bf16(false, a, false, b, (short)0, c,
                                                 false, false);
}

// One pipeline stage of a 16x64 GEMM tile: A frag + 4 B frags (10 b128 loads)
struct Frags {
  v16bf a, b0, b1, b2, b3;
};
__device__ __forceinline__ Frags load_frags(const __bf16* arow,
                                            const __bf16* brow, int k0,
                                            size_t bstride) {
  Frags f;
  f.a  = frag_a(arow + k0);
  f.b0 = frag_b(brow + k0);
  f.b1 = frag_b(brow + bstride + k0);
  f.b2 = frag_b(brow + 2 * bstride + k0);
  f.b3 = frag_b(brow + 3 * bstride + k0);
  return f;
}
__device__ __forceinline__ void mma4(const Frags& f, v8f acc[4]) {
  acc[0] = wmma_bf16(f.a, f.b0, acc[0]);
  acc[1] = wmma_bf16(f.a, f.b1, acc[1]);
  acc[2] = wmma_bf16(f.a, f.b2, acc[2]);
  acc[3] = wmma_bf16(f.a, f.b3, acc[3]);
}
// Per pipeline stage: schedule 10 VMEM reads (next stage) before 4 WMMAs (this stage)
__device__ __forceinline__ void sched_stage() {
  SCHED_GROUP(SG_VMEM_READ, 10, 0);
  SCHED_GROUP(SG_MFMA, 4, 0);
}

// ---------------------------------------------------------------------------
// Kernel 0: f32 -> bf16 conversion
// ---------------------------------------------------------------------------
__global__ void cvt_bf16_kernel(const float* __restrict__ src,
                                __bf16* __restrict__ dst, long long n) {
  long long i = (long long)blockIdx.x * blockDim.x + threadIdx.x;
  if (i < n) dst[i] = (__bf16)src[i];
}

// ---------------------------------------------------------------------------
// Kernel 1: projection GEMM.  C[M x 1024] = X[M x 1024] * W^T + bias
//   mode 0: Q  (M=4096,  X row = input[b, L-1, s]),  store [b][h][s][dh]
//   mode 1: K  (M=20480, X row = (b,l,s)),           store [bl][h][s][dh]
//   mode 2: V  (M=20480),                       store transposed [bl][h][dh][s]
// One wave computes a 16x64 C tile; K-loop software-pipelined (depth 2).
// ---------------------------------------------------------------------------
__global__ __launch_bounds__(256) void proj_kernel(
    const __bf16* __restrict__ X, const __bf16* __restrict__ W,
    const float* __restrict__ bias, __bf16* __restrict__ out, int mode) {
  const int lane    = threadIdx.x & 31;
  const int wid     = blockIdx.x * 8 + (threadIdx.x >> 5);
  const int mt      = wid >> 4;            // 16 N-groups of 64 per M tile
  const int ng      = wid & 15;
  const int m_base  = mt * 16;
  const int n_base  = ng * 64;
  const int mloc    = lane & 15;
  const bool hiHalf = lane >= 16;
  const int koffA   = hiHalf ? 8 : 0;
  const int koffB   = hiHalf ? 8 : 0;

  // A-side source row for this lane
  const int m_row = m_base + mloc;
  int xrow;
  if (mode == 0) {
    const int b = m_row >> 9, s = m_row & (SS - 1);
    xrow = (b * LL + (LL - 1)) * SS + s;          // last lag step only
  } else {
    xrow = m_row;                                  // r encodes (b,l,s)
  }
  const __bf16* arow = X + (size_t)xrow * DD + koffA;
  const __bf16* brow = W + (size_t)(n_base + mloc) * DD + koffB;
  const size_t bstr  = (size_t)16 * DD;

  v8f acc[4] = {};
  // depth-2 software pipeline: load stage i+1 while issuing WMMAs for stage i
  Frags f0 = load_frags(arow, brow, 0, bstr);
  for (int k0 = 0; k0 < DD; k0 += 64) {
    Frags f1 = load_frags(arow, brow, k0 + 32, bstr);
    mma4(f0, acc);
    sched_stage();
    if (k0 + 64 < DD) f0 = load_frags(arow, brow, k0 + 64, bstr);
    mma4(f1, acc);
    sched_stage();
  }

  // epilogue: bias add, bf16 store with per-mode layout
  #pragma unroll
  for (int nf = 0; nf < 4; ++nf) {
    const int col = n_base + nf * 16 + mloc;
    const int hcol = col >> 6, dh = col & 63;
    const float bsv = bias[col];
    #pragma unroll
    for (int j = 0; j < 8; ++j) {
      const int m_out = m_base + j + (hiHalf ? 8 : 0);
      const float v = acc[nf][j] + bsv;
      const int grp = m_out >> 9;            // b (mode 0) or b*L+l (mode 1/2)
      const int ss  = m_out & (SS - 1);
      size_t idx;
      if (mode == 2)
        idx = (((size_t)grp * HH + hcol) * DHH + dh) * SS + ss;   // V^T
      else
        idx = (((size_t)grp * HH + hcol) * SS + ss) * DHH + dh;   // Q / K
      out[idx] = (__bf16)v;
    }
  }
}

// ---------------------------------------------------------------------------
// Kernel 2: flash attention per (b,l,h).  One wave = one 16-query tile.
// Causal + additive (1-mask)*(-1e9) penalty; softmax shift-invariance makes
// this equivalent to the reference's max-subtract-then-penalty ordering.
// The next key tile's K fragments are loaded before the softmax section so
// their latency is hidden behind ~100 VALU ops of online-softmax work.
// ---------------------------------------------------------------------------
__global__ __launch_bounds__(256) void attn_kernel(
    const __bf16* __restrict__ Q,    // [B][H][S][Dh]
    const __bf16* __restrict__ K,    // [B*L][H][S][Dh]
    const __bf16* __restrict__ Vt,   // [B*L][H][Dh][S]
    const int*    __restrict__ am,   // [B][L][S]
    __bf16* __restrict__ inter) {    // [B][S][L*H*Dh]
  __shared__ __align__(16) __bf16 plds[8][16 * 32];

  const int lane    = threadIdx.x & 31;
  const int wslot   = threadIdx.x >> 5;
  const int wid     = blockIdx.x * 8 + wslot;
  const int qt      = wid & 31;            // 32 query tiles of 16
  const int blh     = wid >> 5;            // (b*L + l)*H + h
  const int h       = blh & (HH - 1);
  const int bl      = blh >> 4;
  const int lg      = bl % LL;
  const int b       = bl / LL;
  const int qm      = qt * 16;
  const int mloc    = lane & 15;
  const bool hiHalf = lane >= 16;
  const int koffA   = hiHalf ? 8 : 0;
  const int koffB   = hiHalf ? 8 : 0;

  const __bf16* qrow  = Q  + ((((size_t)b * HH + h) * SS) + (qm + mloc)) * DHH;
  const __bf16* Kb    = K  + (((size_t)bl * HH + h) * SS) * DHH;
  const __bf16* Vb    = Vt + (((size_t)bl * HH + h) * DHH) * SS;
  const int*    amrow = am + ((size_t)b * LL + lg) * SS;

  const v16bf qa0 = frag_a(qrow + 0  + koffA);   // Q tile, k = 0..31
  const v16bf qa1 = frag_a(qrow + 32 + koffA);   // Q tile, k = 32..63

  float m_run[8], l_run[8];
  v8f o[4] = {};
  #pragma unroll
  for (int j = 0; j < 8; ++j) { m_run[j] = -3.0e38f; l_run[j] = 0.0f; }

  auto load_kf = [&](int kb, v16bf kf[4]) {
    const __bf16* kp = Kb + (size_t)(kb + mloc) * DHH + koffB;
    kf[0] = frag_b(kp);
    kf[1] = frag_b(kp + 32);
    kf[2] = frag_b(kp + 16 * DHH);
    kf[3] = frag_b(kp + 16 * DHH + 32);
  };

  const int nkt = (qt + 2) >> 1;                 // ceil((qm+16)/32) key steps
  v16bf kf[4];
  load_kf(0, kf);
  for (int kt = 0; kt < nkt; ++kt) {
    const int kb = kt * 32;

    // V fragments for this tile (independent of the scores)
    v16bf vf[4];
    #pragma unroll
    for (int nf = 0; nf < 4; ++nf)
      vf[nf] = frag_b(Vb + (size_t)(nf * 16 + mloc) * SS + kb + koffB);

    // ---- S = Q K^T / sqrt(Dh) for 32 keys (two 16x16 score frags) ----
    v8f s0 = {}, s1 = {};
    s0 = wmma_bf16(qa0, kf[0], s0);
    s0 = wmma_bf16(qa1, kf[1], s0);
    s1 = wmma_bf16(qa0, kf[2], s1);
    s1 = wmma_bf16(qa1, kf[3], s1);

    // prefetch next key tile's K fragments; latency hidden by softmax below
    if (kt + 1 < nkt) load_kf(kb + 32, kf);

    const int key0 = kb + mloc, key1 = kb + 16 + mloc;
    const float pen0 = (1.0f - (float)amrow[key0]) * (-1e9f);
    const float pen1 = (1.0f - (float)amrow[key1]) * (-1e9f);

    // ---- online softmax update ----
    #pragma unroll
    for (int j = 0; j < 8; ++j) {
      const int qg = qm + j + (hiHalf ? 8 : 0);
      float x0 = s0[j] * 0.125f + pen0;
      float x1 = s1[j] * 0.125f + pen1;
      if (key0 > qg) x0 = -3.0e38f;
      if (key1 > qg) x1 = -3.0e38f;
      float mt = fmaxf(x0, x1);
      #pragma unroll
      for (int d = 1; d < 16; d <<= 1) mt = fmaxf(mt, __shfl_xor(mt, d, 32));
      const float mnew = fmaxf(m_run[j], mt);
      const float al   = __expf(m_run[j] - mnew);
      const float p0   = __expf(x0 - mnew);
      const float p1   = __expf(x1 - mnew);
      float sm = p0 + p1;
      #pragma unroll
      for (int d = 1; d < 16; d <<= 1) sm += __shfl_xor(sm, d, 32);
      l_run[j] = l_run[j] * al + sm;
      m_run[j] = mnew;
      #pragma unroll
      for (int nf = 0; nf < 4; ++nf) o[nf][j] *= al;
      const int mr = j + (hiHalf ? 8 : 0);
      plds[wslot][mr * 32 + mloc]      = (__bf16)p0;   // C-frag -> LDS tile
      plds[wslot][mr * 32 + 16 + mloc] = (__bf16)p1;
    }

    // wave-private LDS tile: order writes before the A-fragment gather
    asm volatile("s_wait_dscnt 0" ::: "memory");

    const v16bf pf = frag_a(&plds[wslot][mloc * 32 + koffA]);  // P as A-frag

    // ---- O += P * V  (V^T rows are contiguous in key) ----
    o[0] = wmma_bf16(pf, vf[0], o[0]);
    o[1] = wmma_bf16(pf, vf[1], o[1]);
    o[2] = wmma_bf16(pf, vf[2], o[2]);
    o[3] = wmma_bf16(pf, vf[3], o[3]);
  }

  // ---- finalize: O /= l, scatter into inter[b][s][(l*H+h)*Dh + dh] ----
  #pragma unroll
  for (int j = 0; j < 8; ++j) {
    const float inv = 1.0f / l_run[j];
    const int qv = qm + j + (hiHalf ? 8 : 0);
    const size_t rbase =
        ((size_t)b * SS + qv) * LD_O + ((size_t)lg * HH + h) * DHH;
    #pragma unroll
    for (int nf = 0; nf < 4; ++nf)
      inter[rbase + nf * 16 + mloc] = (__bf16)(o[nf][j] * inv);
  }
}

// ---------------------------------------------------------------------------
// Kernel 3: output projection.  out[4096 x 1024] = inter[4096 x 5120] * Wo^T + bo
// ---------------------------------------------------------------------------
__global__ __launch_bounds__(256) void outproj_kernel(
    const __bf16* __restrict__ A,   // [4096][5120]
    const __bf16* __restrict__ W,   // [1024][5120]
    const float* __restrict__ bias, float* __restrict__ out) {
  const int lane    = threadIdx.x & 31;
  const int wid     = blockIdx.x * 8 + (threadIdx.x >> 5);
  const int mt      = wid >> 4;
  const int ng      = wid & 15;
  const int m_base  = mt * 16;
  const int n_base  = ng * 64;
  const int mloc    = lane & 15;
  const bool hiHalf = lane >= 16;
  const int koffA   = hiHalf ? 8 : 0;
  const int koffB   = hiHalf ? 8 : 0;

  const __bf16* arow = A + (size_t)(m_base + mloc) * LD_O + koffA;
  const __bf16* brow = W + (size_t)(n_base + mloc) * LD_O + koffB;
  const size_t bstr  = (size_t)16 * LD_O;

  v8f acc[4] = {};
  Frags f0 = load_frags(arow, brow, 0, bstr);
  for (int k0 = 0; k0 < LD_O; k0 += 64) {
    Frags f1 = load_frags(arow, brow, k0 + 32, bstr);
    mma4(f0, acc);
    sched_stage();
    if (k0 + 64 < LD_O) f0 = load_frags(arow, brow, k0 + 64, bstr);
    mma4(f1, acc);
    sched_stage();
  }

  #pragma unroll
  for (int nf = 0; nf < 4; ++nf) {
    const int col = n_base + nf * 16 + mloc;
    const float bsv = bias[col];
    #pragma unroll
    for (int j = 0; j < 8; ++j) {
      const int m_out = m_base + j + (hiHalf ? 8 : 0);
      out[(size_t)m_out * DD + col] = acc[nf][j] + bsv;
    }
  }
}

// ---------------------------------------------------------------------------
// Host launcher.  Workspace layout (~193 MB):
//   Xbf(40M) | Wq(2M) | Wk(2M) | Wv(2M) | Wo(10M) | Q(8M) | K(40M) | Vt(40M)
//   | inter(40M)   (bf16 buffers, 256-byte aligned)
// ---------------------------------------------------------------------------
extern "C" void kernel_launch(void* const* d_in, const int* in_sizes, int n_in,
                              void* d_out, int out_size, void* d_ws,
                              size_t ws_size, hipStream_t stream) {
  const float* input = (const float*)d_in[0];
  const int*   amask = (const int*)d_in[1];
  const float* Wq    = (const float*)d_in[2];
  const float* bq    = (const float*)d_in[3];
  const float* Wk    = (const float*)d_in[4];
  const float* bk    = (const float*)d_in[5];
  const float* Wv    = (const float*)d_in[6];
  const float* bv    = (const float*)d_in[7];
  const float* Wo    = (const float*)d_in[8];
  const float* bo    = (const float*)d_in[9];
  float* out = (float*)d_out;

  char* ws = (char*)d_ws;
  size_t off = 0;
  auto take = [&](size_t nelem) -> __bf16* {
    __bf16* p = (__bf16*)(ws + off);
    off += ((nelem * sizeof(__bf16)) + 255) & ~(size_t)255;
    return p;
  };
  const size_t nX  = (size_t)BB * LL * SS * DD;   // 20,971,520
  const size_t nWp = (size_t)DD * DD;             // 1,048,576
  const size_t nWo = (size_t)DD * LD_O;           // 5,242,880
  const size_t nQ  = (size_t)BB * SS * DD;        // 4,194,304
  const size_t nI  = (size_t)BB * SS * LD_O;      // 20,971,520

  __bf16* Xbf   = take(nX);
  __bf16* Wqb   = take(nWp);
  __bf16* Wkb   = take(nWp);
  __bf16* Wvb   = take(nWp);
  __bf16* Wob   = take(nWo);
  __bf16* Qbf   = take(nQ);
  __bf16* Kbf   = take(nX);
  __bf16* Vtb   = take(nX);
  __bf16* Inter = take(nI);

  auto cvt = [&](const float* s, __bf16* d, size_t n) {
    cvt_bf16_kernel<<<(unsigned)((n + 255) / 256), 256, 0, stream>>>(
        s, d, (long long)n);
  };
  cvt(input, Xbf, nX);
  cvt(Wq, Wqb, nWp);
  cvt(Wk, Wkb, nWp);
  cvt(Wv, Wvb, nWp);
  cvt(Wo, Wob, nWo);

  // Q: 4096/16 * 16 = 4096 waves -> 512 blocks.  K/V: 20480 waves -> 2560.
  proj_kernel<<<512, 256, 0, stream>>>(Xbf, Wqb, bq, Qbf, 0);
  proj_kernel<<<2560, 256, 0, stream>>>(Xbf, Wkb, bk, Kbf, 1);
  proj_kernel<<<2560, 256, 0, stream>>>(Xbf, Wvb, bv, Vtb, 2);

  // attention: B*L*H*32 = 20480 waves -> 2560 blocks
  attn_kernel<<<2560, 256, 0, stream>>>(Qbf, Kbf, Vtb, amask, Inter);

  // output projection: 4096 waves -> 512 blocks
  outproj_kernel<<<512, 256, 0, stream>>>(Inter, Wob, bo, out);
}
